// BootDecoder_721554506545
// MI455X (gfx1250) — matrix operand order repaired and checked by reading, not compiled, not run on hardware
//
#include <hip/hip_runtime.h>
#include <math.h>

typedef __attribute__((ext_vector_type(16))) _Float16 v16h;
typedef __attribute__((ext_vector_type(8)))  float    v8f;

#define CC 8
#define FF 256
#define SEEDC 16
#define TS 260   // padded LDS row stride (floats) to dodge bank conflicts

// ---------------- init: zero masks + hx ----------------
__global__ __launch_bounds__(256)
void init_kernel(unsigned* __restrict__ mask, float* __restrict__ hx, int n) {
    int i = blockIdx.x * blockDim.x + threadIdx.x;
    if (i < n) mask[i] = 0u;
    if (i < CC * FF) hx[i] = 0.0f;
}

__global__ __launch_bounds__(128)
void seed_kernel(const int* __restrict__ seeds, unsigned* __restrict__ mask,
                 int* __restrict__ last_sel) {
    int p = threadIdx.x;              // p = c*16 + s
    int node = seeds[p];
    atomicOr(&mask[node], (1u << (p >> 4)) | 0x100u);  // class bit + entity bit
    last_sel[p] = node;
}

__global__ __launch_bounds__(256)
void zero_u32(unsigned* __restrict__ p, int n) {
    int i = blockIdx.x * blockDim.x + threadIdx.x;
    if (i < n) p[i] = 0u;
}

// ---------------- precompute normalized-es f16 A-fragments (step-invariant) --------
__global__ __launch_bounds__(32)
void esfrag_kernel(const float* __restrict__ es, _Float16* __restrict__ esfrag, int n) {
    __shared__ float tile[16 * TS];
    __shared__ float rinv[16];
    const int lane = threadIdx.x;
    const int t0   = blockIdx.x;
    const int n0   = t0 * 16;
    if (n0 + 32 <= n) __builtin_prefetch(es + (size_t)(n0 + 16) * FF, 0, 1);
    for (int r = 0; r < 16; ++r) {
        const float* src = es + (size_t)(n0 + r) * FF;
        for (int k = lane; k < FF; k += 32) tile[r * TS + k] = src[k];
    }
    __syncthreads();
    if (lane < 16) {
        float s = 0.0f;
        for (int k = 0; k < FF; ++k) { float v = tile[lane * TS + k]; s += v * v; }
        rinv[lane] = 1.0f / (sqrtf(s) + 1e-8f);
    }
    __syncthreads();
    const int half = lane >> 4;
    const int M    = lane & 15;
    const float rv = rinv[M];
    for (int kb = 0; kb < 8; ++kb) {
        v16h a;
        #pragma unroll
        for (int j = 0; j < 8; ++j) {                 // A 16x32: VGPR0-3 -> K = 8*half + j
            int K = kb * 32 + half * 8 + j;
            a[j] = (_Float16)(tile[M * TS + K] * rv);
        }
        #pragma unroll
        for (int j = 0; j < 8; ++j) {                 // VGPR4-7 -> K = 16 + 8*half + j
            int K = kb * 32 + 16 + half * 8 + j;
            a[8 + j] = (_Float16)(tile[M * TS + K] * rv);
        }
        *(v16h*)(esfrag + (size_t)(((size_t)t0 * 8 + kb) * 32 + lane) * 16) = a;
    }
}

// ---------------- GRU (one block per class) ----------------
__global__ __launch_bounds__(256)
void gru_kernel(const float* __restrict__ es, const int* __restrict__ last_sel,
                const float* __restrict__ W_ih, const float* __restrict__ W_hh,
                const float* __restrict__ b_ih, const float* __restrict__ b_hh,
                float* __restrict__ hx, float* __restrict__ out_hx) {
    __shared__ float inp[FF];
    __shared__ float hsh[FF];
    const int c = blockIdx.x;
    const int f = threadIdx.x;
    float s = 0.0f;
    for (int j = 0; j < SEEDC; ++j) {
        int idx = last_sel[c * SEEDC + j];
        s += es[(size_t)idx * FF + f];
    }
    inp[f] = s * (1.0f / 16.0f);
    hsh[f] = hx[c * FF + f];
    __syncthreads();

    float ir = b_ih[f],        iz = b_ih[FF + f],        in_ = b_ih[2 * FF + f];
    float hr = b_hh[f],        hz = b_hh[FF + f],        hn  = b_hh[2 * FF + f];
    const float* wir = W_ih + (size_t)f * FF;
    const float* wiz = W_ih + (size_t)(FF + f) * FF;
    const float* win = W_ih + (size_t)(2 * FF + f) * FF;
    const float* whr = W_hh + (size_t)f * FF;
    const float* whz = W_hh + (size_t)(FF + f) * FF;
    const float* whn = W_hh + (size_t)(2 * FF + f) * FF;
    for (int k = 0; k < FF; ++k) {
        float x = inp[k], h = hsh[k];
        ir  += wir[k] * x;  iz += wiz[k] * x;  in_ += win[k] * x;
        hr  += whr[k] * h;  hz += whz[k] * h;  hn  += whn[k] * h;
    }
    float r = 1.0f / (1.0f + expf(-(ir + hr)));
    float z = 1.0f / (1.0f + expf(-(iz + hz)));
    float nn = tanhf(in_ + r * hn);
    float hnew = (1.0f - z) * nn + z * hsh[f];
    hx[c * FF + f] = hnew;
    out_hx[c * FF + f] = hnew;
}

// ---------------- build normalized-hx B fragments (f16, WMMA 32x16 layout) ----------------
__global__ __launch_bounds__(32)
void prep_b_kernel(const float* __restrict__ hx, _Float16* __restrict__ bfrag) {
    __shared__ float rinv[16];
    const int lane = threadIdx.x;
    if (lane < 16) {
        float rv = 0.0f;
        if (lane < CC) {
            float s = 0.0f;
            for (int k = 0; k < FF; ++k) { float v = hx[lane * FF + k]; s += v * v; }
            rv = 1.0f / (sqrtf(s) + 1e-8f);
        }
        rinv[lane] = rv;
    }
    __syncthreads();
    const int half = lane >> 4;
    const int col  = lane & 15;
    for (int kb = 0; kb < 8; ++kb) {
        for (int j = 0; j < 16; ++j) {
            int K = kb * 32 + (j < 8 ? half * 8 + j : 16 + half * 8 + (j - 8));  // B 32x16: j<8 -> K=8*half+j, j>=8 -> K=16+8*half+(j-8)
            float v = (col < CC) ? hx[col * FF + K] * rinv[col] : 0.0f;
            bfrag[(size_t)(kb * 32 + lane) * 16 + j] = (_Float16)v;
        }
    }
}

// ---------------- edge count (segment_sum of class membership) ----------------
__global__ __launch_bounds__(256)
void edge_kernel(const int* __restrict__ esrc, const int* __restrict__ edst,
                 const unsigned* __restrict__ mask, unsigned* __restrict__ counts, int e) {
    int i = blockIdx.x * blockDim.x + threadIdx.x;
    if (i >= e) return;
    int dst = edst[i];
    unsigned m = mask[dst] & 0xFFu;
    if (!m) return;                                   // fast path: ~99.97% of edges
    int src = esrc[i];
    while (m) {
        int c = __ffs((int)m) - 1;
        atomicAdd(&counts[(size_t)src * CC + c], 1u);
        m &= m - 1;
    }
}

// -------- shared tail: validity / probs / sparse candidate append ----------------
__device__ __forceinline__
void emit_results(const v8f& acc, int n0, int half, int col, int mm, int n,
                  const unsigned* __restrict__ mask, const unsigned* __restrict__ counts,
                  float* __restrict__ probs, float* __restrict__ cand_val,
                  int* __restrict__ cand_idx, unsigned* __restrict__ cand_cnt) {
    #pragma unroll
    for (int r = 0; r < 8; ++r) {
        int node = n0 + r + 8 * half;                 // D: VGPR r -> row r + 8*half
        unsigned mw = mask[node];
        bool entity = (mw >> 8) & 1u;
        bool valid = false;
        if (col < CC) {
            unsigned cnt = counts[(size_t)node * CC + col];
            valid = (cnt >= (unsigned)mm) && !entity;
        }
        unsigned bal = __builtin_amdgcn_ballot_w32(valid);
        bool pools = half ? ((bal & 0x00FF0000u) != 0u) : ((bal & 0x000000FFu) != 0u);
        if (col < CC) {
            float prob = pools ? (0.5f * acc[r] + 0.5f) : 0.0f;
            probs[(size_t)node * CC + col] = prob;
            if (valid) {                              // sparse: expected O(10) per step
                unsigned pos = atomicAdd(&cand_cnt[col], 1u);
                cand_val[(size_t)col * n + pos] = prob;
                cand_idx[(size_t)col * n + pos] = node;
            }
        }
    }
}

// ---------------- FAST sim: B frags in regs, staged A loads, 4 tiles/wave -------------
__global__ __launch_bounds__(256)
void sim_fast_kernel(const _Float16* __restrict__ esfrag, const _Float16* __restrict__ bfrag,
                     const unsigned* __restrict__ mask, const unsigned* __restrict__ counts,
                     float* __restrict__ probs, float* __restrict__ cand_val,
                     int* __restrict__ cand_idx, unsigned* __restrict__ cand_cnt,
                     int mm, int n, int ntiles) {
    const int wave = threadIdx.x >> 5;
    const int lane = threadIdx.x & 31;
    const int tileBase = (blockIdx.x * 8 + wave) * 4;
    if (tileBase >= ntiles) return;

    v16h b[8];                                        // all 8 B fragments resident (64 VGPRs)
    #pragma unroll
    for (int kb = 0; kb < 8; ++kb)
        b[kb] = *(const v16h*)(bfrag + (size_t)(kb * 32 + lane) * 16);

    const int half = lane >> 4;
    const int col  = lane & 15;
    for (int t = 0; t < 4; ++t) {
        const int tile = tileBase + t;
        if (tile >= ntiles) break;
        if (tile + 1 < ntiles)
            __builtin_prefetch(esfrag + (size_t)(tile + 1) * 8 * 32 * 16, 0, 1);
        const _Float16* ef = esfrag + (size_t)tile * 8 * 32 * 16;

        // stage ALL 8 A fragments into distinct registers: one clause of 16 b128 loads
        // in flight, then drain with staged s_wait_loadcnt between WMMAs
        v16h a[8];
        #pragma unroll
        for (int kb = 0; kb < 8; ++kb)
            a[kb] = *(const v16h*)(ef + (size_t)(kb * 32 + lane) * 16);

        v8f acc = {};
        #pragma unroll
        for (int kb = 0; kb < 8; ++kb)
            acc = __builtin_amdgcn_wmma_f32_16x16x32_f16(false, a[kb], false, b[kb],
                                                         (short)0, acc, false, false);
        emit_results(acc, tile * 16, half, col, mm, n, mask, counts,
                     probs, cand_val, cand_idx, cand_cnt);
    }
}

// ---------------- fallback sim (LDS staging, used only if ws too small) ----------------
__global__ __launch_bounds__(32)
void sim_kernel(const float* __restrict__ es, const _Float16* __restrict__ bfrag,
                const unsigned* __restrict__ mask, const unsigned* __restrict__ counts,
                float* __restrict__ probs, float* __restrict__ cand_val,
                int* __restrict__ cand_idx, unsigned* __restrict__ cand_cnt,
                int mm, int n) {
    __shared__ float tile[16 * TS];
    __shared__ float rinv[16];
    const int lane = threadIdx.x;
    const int n0   = blockIdx.x * 16;
    if (n0 + 32 <= n) __builtin_prefetch(es + (size_t)(n0 + 16) * FF, 0, 1);
    for (int r = 0; r < 16; ++r) {
        const float* src = es + (size_t)(n0 + r) * FF;
        for (int k = lane; k < FF; k += 32) tile[r * TS + k] = src[k];
    }
    __syncthreads();
    if (lane < 16) {
        float s = 0.0f;
        for (int k = 0; k < FF; ++k) { float v = tile[lane * TS + k]; s += v * v; }
        rinv[lane] = 1.0f / (sqrtf(s) + 1e-8f);
    }
    __syncthreads();
    const int half = lane >> 4;
    const int M    = lane & 15;
    const float rv = rinv[M];
    v8f acc = {};
    for (int kb = 0; kb < 8; ++kb) {
        v16h a;
        #pragma unroll
        for (int j = 0; j < 8; ++j) {
            int K = kb * 32 + half * 8 + j;
            a[j] = (_Float16)(tile[M * TS + K] * rv);
        }
        #pragma unroll
        for (int j = 0; j < 8; ++j) {
            int K = kb * 32 + 16 + half * 8 + j;
            a[8 + j] = (_Float16)(tile[M * TS + K] * rv);
        }
        v16h b = *(const v16h*)(bfrag + (size_t)(kb * 32 + lane) * 16);
        acc = __builtin_amdgcn_wmma_f32_16x16x32_f16(false, a, false, b,
                                                     (short)0, acc, false, false);
    }
    emit_results(acc, n0, half, lane & 15, mm, n, mask, counts,
                 probs, cand_val, cand_idx, cand_cnt);
}

// ------- per-class top-16 over the sparse candidate list (jax tie semantics) ----------
__global__ __launch_bounds__(256)
void topk_kernel(float* __restrict__ cand_val, const int* __restrict__ cand_idx,
                 const unsigned* __restrict__ cand_cnt,
                 const unsigned* __restrict__ counts, const unsigned* __restrict__ mask,
                 int* __restrict__ sel, int mm, int n) {
    __shared__ float bv[256];
    __shared__ int   bi[256];
    __shared__ int   bp[256];
    __shared__ int   fcur;
    const int c = blockIdx.x, t = threadIdx.x;
    int V = (int)cand_cnt[c]; if (V > n) V = n;
    float*     cv = cand_val + (size_t)c * n;
    const int* ci = cand_idx + (size_t)c * n;
    if (t == 0) fcur = 0;
    __syncthreads();
    for (int it = 0; it < SEEDC; ++it) {
        float best = -1e30f; int bidx = n; int bpos = -1;
        for (int i = t; i < V; i += 256) {
            float v = cv[i]; int id = ci[i];
            if (v > best || (v == best && id < bidx)) { best = v; bidx = id; bpos = i; }
        }
        bv[t] = best; bi[t] = bidx; bp[t] = bpos;
        __syncthreads();
        for (int off = 128; off > 0; off >>= 1) {
            if (t < off) {
                float ov = bv[t + off]; int oi = bi[t + off];
                if (ov > bv[t] || (ov == bv[t] && oi < bi[t])) {
                    bv[t] = ov; bi[t] = oi; bp[t] = bp[t + off];
                }
            }
            __syncthreads();
        }
        if (t == 0) {
            if (bi[0] < n) {                           // live candidate remains
                sel[c * SEEDC + it] = bi[0];
                cv[bp[0]] = -2e30f;                    // remove from later rounds
            } else {                                   // jax -1-tie filler: smallest non-valid index
                int f = fcur;
                while (true) {
                    bool entity = (mask[f] >> 8) & 1u;
                    bool valid = (counts[(size_t)f * CC + c] >= (unsigned)mm) && !entity;
                    if (!valid) break;
                    ++f;
                }
                sel[c * SEEDC + it] = f;
                fcur = f + 1;
            }
        }
        __threadfence();
        __syncthreads();
    }
}

// ---------------- apply selections + emit outputs/selects/steps ----------------
__global__ __launch_bounds__(128)
void update_kernel(const int* __restrict__ sel, const float* __restrict__ probs,
                   unsigned* __restrict__ mask, int* __restrict__ last_sel,
                   float* __restrict__ out, int step) {
    int t = threadIdx.x;               // t = c*16 + s, same order as group.reshape(-1)
    int c = t >> 4;
    int idx = sel[t];
    atomicOr(&mask[idx], (1u << c) | 0x100u);
    last_sel[t] = idx;
    for (int cc = 0; cc < CC; ++cc)
        out[step * 1024 + t * CC + cc] = probs[(size_t)idx * CC + cc];  // outputs [4,128,8]
    out[4096 + step * 128 + t] = (float)idx;                            // selects [4,128]
    if (t < CC) out[4608 + step * CC + t] = 16.0f;                      // steps   [4,8]
}

// ---------------- host-side orchestration ----------------
extern "C" void kernel_launch(void* const* d_in, const int* in_sizes, int n_in,
                              void* d_out, int out_size, void* d_ws, size_t ws_size,
                              hipStream_t stream) {
    const int*   seeds = (const int*)  d_in[0];
    const int*   esrc  = (const int*)  d_in[1];
    const int*   edst  = (const int*)  d_in[2];
    const float* es    = (const float*)d_in[3];
    const float* W_ih  = (const float*)d_in[4];
    const float* W_hh  = (const float*)d_in[5];
    const float* b_ih  = (const float*)d_in[6];
    const float* b_hh  = (const float*)d_in[7];
    float* out = (float*)d_out;

    const int E = in_sizes[1];
    const int N = in_sizes[3] / FF;
    const int NT = N / 16;             // 12500 tiles

    // workspace carve-up (256B aligned); esfrag last so fallback layout still fits
    char* ws = (char*)d_ws;
    size_t off = 0;
    auto take = [&](size_t bytes) { char* p = ws + off; off = (off + bytes + 255) & ~(size_t)255; return p; };
    float*    hx       = (float*)   take((size_t)CC * FF * 4);
    unsigned* mask     = (unsigned*)take((size_t)N * 4);
    int*      last_sel = (int*)     take((size_t)CC * SEEDC * 4);
    unsigned* counts   = (unsigned*)take((size_t)N * CC * 4);
    float*    probs    = (float*)   take((size_t)N * CC * 4);
    float*    cand_val = (float*)   take((size_t)CC * N * 4);
    int*      cand_idx = (int*)     take((size_t)CC * N * 4);
    unsigned* cand_cnt = (unsigned*)take((size_t)CC * 4);
    _Float16* bfrag    = (_Float16*)take((size_t)8 * 32 * 16 * 2);
    int*      sel      = (int*)     take((size_t)CC * SEEDC * 4);
    _Float16* esfrag   = (_Float16*)take((size_t)N * FF * 2);   // 102 MB fragment cache
    const bool use_fast = (off <= ws_size);
    (void)n_in; (void)out_size;

    init_kernel<<<(N + 255) / 256, 256, 0, stream>>>(mask, hx, N);
    seed_kernel<<<1, 128, 0, stream>>>(seeds, mask, last_sel);
    if (use_fast)
        esfrag_kernel<<<NT, 32, 0, stream>>>(es, esfrag, N);

    for (int step = 0; step < 4; ++step) {
        int mm = (step == 0) ? 3 : 2;   // max(2, MIN_MATCH - step)
        gru_kernel<<<CC, FF, 0, stream>>>(es, last_sel, W_ih, W_hh, b_ih, b_hh,
                                          hx, out + 4640 + step * (CC * FF));
        prep_b_kernel<<<1, 32, 0, stream>>>(hx, bfrag);
        zero_u32<<<(N * CC + 255) / 256, 256, 0, stream>>>(counts, N * CC);
        zero_u32<<<1, 256, 0, stream>>>(cand_cnt, CC);
        edge_kernel<<<(E + 255) / 256, 256, 0, stream>>>(esrc, edst, mask, counts, E);
        if (use_fast)
            sim_fast_kernel<<<(NT / 4 + 7) / 8, 256, 0, stream>>>(esfrag, bfrag, mask, counts,
                                                                  probs, cand_val, cand_idx,
                                                                  cand_cnt, mm, N, NT);
        else
            sim_kernel<<<NT, 32, 0, stream>>>(es, bfrag, mask, counts, probs,
                                              cand_val, cand_idx, cand_cnt, mm, N);
        topk_kernel<<<CC, 256, 0, stream>>>(cand_val, cand_idx, cand_cnt,
                                            counts, mask, sel, mm, N);
        update_kernel<<<1, 128, 0, stream>>>(sel, probs, mask, last_sel, out, step);
    }
}
